// Attention_Layer_3599182594080
// MI455X (gfx1250) — matrix operand order repaired and checked by reading, not compiled, not run on hardware
//
#include <hip/hip_runtime.h>

// ---------------------------------------------------------------------------
// Graph-attention layer for MI455X (gfx1250, wave32, WMMA).
// Shapes: B=4, N=512, D=128, DH=64.
// Pipeline:
//   k1: Q,K,V,Ni,Nj = h @ W^T + b                       (tiny, VALU)
//   k2: E = e @ We^T (f16 WMMA, f32 accum), fused:
//         e_out = E + be + Ni[i] + Nj[j]  (NT-streamed to d_out)
//         attT[b,j,i] = (1/8) * sum_d e_out * Q[i,d] * K[j,d]
//   k3: softmax over i (axis 1) per (b,j) column
//   k4: h_out = att @ V                                  (tiny, VALU)
// Roofline: k2 moves ~768MB (e read + e_out write) -> ~33us @ 23.3 TB/s;
// f16-in/f32-acc WMMA keeps the 17.2 GFLOP GEMM under the HBM roofline.
// ---------------------------------------------------------------------------

typedef __attribute__((ext_vector_type(16))) _Float16 v16h;
typedef __attribute__((ext_vector_type(8)))  _Float16 v8h;
typedef __attribute__((ext_vector_type(8)))  float    v8f;
typedef __attribute__((ext_vector_type(4)))  float    v4f;

#define GB   4
#define GN   512
#define GD   128
#define GDH  64
#define ROWS (GB * GN * GN)          // 1,048,576 flattened (b,i,j) rows
#define WPITCH 136                   // LDS pitch (halves) for We, bank-spread
#define REPS 4                       // tile-groups per block (amortize We stage)

// ---------------------------------------------------------------------------
// Kernel 1: five row-projections of h. out = [Q | K | V | Ni | Nj], each
// [B*N, 64] f32, contiguous.
// ---------------------------------------------------------------------------
__global__ __launch_bounds__(320) void proj_kernel(
    const float* __restrict__ h,
    const float* __restrict__ Wq,  const float* __restrict__ bq,
    const float* __restrict__ Wk,  const float* __restrict__ bk,
    const float* __restrict__ Wv,  const float* __restrict__ bv,
    const float* __restrict__ Wni, const float* __restrict__ bni,
    const float* __restrict__ Wnj, const float* __restrict__ bnj,
    float* __restrict__ out)
{
    __shared__ float hrow[GD];
    const int row = blockIdx.x;          // 0..B*N-1
    const int t   = threadIdx.x;         // 0..319
    if (t < GD) hrow[t] = h[row * GD + t];
    __syncthreads();

    const int w = t >> 6;                // which weight matrix
    const int d = t & 63;
    const float* W; const float* bb;
    switch (w) {
        case 0:  W = Wq;  bb = bq;  break;
        case 1:  W = Wk;  bb = bk;  break;
        case 2:  W = Wv;  bb = bv;  break;
        case 3:  W = Wni; bb = bni; break;
        default: W = Wnj; bb = bnj; break;
    }
    float acc = bb[d];
    const float* wr = W + d * GD;
    #pragma unroll 8
    for (int k = 0; k < GD; ++k) acc += hrow[k] * wr[k];
    out[(long)w * (GB * GN * GDH) + (long)row * GDH + d] = acc;
}

// ---------------------------------------------------------------------------
// Kernel 2: main fused GEMM. One wave per 16-row tile; 8 waves/block; each
// block runs REPS tile-groups. grid = ROWS/16/8/REPS = 2048 blocks.
// ---------------------------------------------------------------------------
__global__ __launch_bounds__(256) void egemm_kernel(
    const float* __restrict__ e,
    const float* __restrict__ We, const float* __restrict__ be,
    const float* __restrict__ Q,  const float* __restrict__ K,
    const float* __restrict__ Ni, const float* __restrict__ Nj,
    float* __restrict__ e_out,    // [ROWS, 64]
    float* __restrict__ attT)     // [B, N(j), N(i)]
{
    __shared__ _Float16 Wlds[64 * WPITCH];
    __shared__ float    belds[64];

    const int t = threadIdx.x;
    // Stage We (64x128 f32) into LDS as f16, padded pitch (once per block).
    for (int idx = t; idx < 64 * GD; idx += 256) {
        const int r = idx >> 7, c = idx & 127;
        Wlds[r * WPITCH + c] = (_Float16)We[idx];
    }
    if (t < 64) belds[t] = be[t];
    __syncthreads();

    const int lane = t & 31;
    const int wave = t >> 5;
    const int m    = lane & 15;          // row (A) / col (B,C) index
    const int half = lane >> 4;          // half-wave select

    for (int rep = 0; rep < REPS; ++rep) {
        const long tile = ((long)blockIdx.x * REPS + rep) * 8 + wave;
        const long r0   = tile * 16;                    // first flattened row
        const int  j0   = (int)(r0 & (GN - 1));         // tile-aligned j base
        const int  i    = (int)((r0 >> 9) & (GN - 1));
        const int  b    = (int)(r0 >> 18);
        const int  bi   = b * GN + i;

        // Prefetch next rep's A rows (overlap DRAM with this rep's compute).
        if (rep < REPS - 1) {
            const float* nrow = e + (r0 + 8 * 16 + m) * GD;   // +8 tiles ahead
            __builtin_prefetch(nrow + half * 64, 0, 3);        // lines 0,2
            __builtin_prefetch(nrow + 32 + half * 64, 0, 3);   // lines 1,3
        }

        v8f c[4] = {v8f{}, v8f{}, v8f{}, v8f{}};

        const float* arow = e + (r0 + m) * GD;          // this lane's A row
        #pragma unroll
        for (int kb = 0; kb < GD; kb += 32) {
            // ---- A fragment: 16-bit A 16x32 layout, 8-element K chunks ----
            v16h a;
            {
                const float* p0 = arow + kb + half * 8;        // K chunk 0
                const float* p1 = arow + kb + 16 + half * 8;   // K chunk +16
                const v4f f0 = *(const v4f*)(p0);
                const v4f f1 = *(const v4f*)(p0 + 4);
                const v4f f2 = *(const v4f*)(p1);
                const v4f f3 = *(const v4f*)(p1 + 4);
                #pragma unroll
                for (int q = 0; q < 4; ++q) {
                    a[q]      = (_Float16)f0[q];
                    a[q + 4]  = (_Float16)f1[q];
                    a[q + 8]  = (_Float16)f2[q];
                    a[q + 12] = (_Float16)f3[q];
                }
            }
            // ---- 4 N-tiles of 16 columns each ----
            #pragma unroll
            for (int nt = 0; nt < 4; ++nt) {
                // B 32x16: lanes 0-15 hold K=0..7,16..23; lanes 16-31 K=8..15,24..31
                const _Float16* wp =
                    &Wlds[(nt * 16 + m) * WPITCH + kb + half * 8];
                const v8h blo = *(const v8h*)(wp);
                const v8h bhi = *(const v8h*)(wp + 16);
                v16h bf;
                #pragma unroll
                for (int q = 0; q < 8; ++q) { bf[q] = blo[q]; bf[q + 8] = bhi[q]; }
                c[nt] = __builtin_amdgcn_wmma_f32_16x16x32_f16(
                    false, a, false, bf, (short)0, c[nt], false, false);
            }
        }

        // ---- Fused epilogue: bias + Ni + Nj, NT-store e_out, att dots ----
        const int mb = half * 8;              // C rows: lanes>=16 hold M=8..15
        float pv[8] = {0.f, 0.f, 0.f, 0.f, 0.f, 0.f, 0.f, 0.f};
        const float* Qrow   = Q  + (long)bi * GDH;
        const float* Nirow  = Ni + (long)bi * GDH;
        const float* Krow0  = K  + (long)(b * GN + j0) * GDH;
        const float* Njrow0 = Nj + (long)(b * GN + j0) * GDH;

        #pragma unroll
        for (int nt = 0; nt < 4; ++nt) {
            const int   n      = nt * 16 + m;         // this lane's column
            const float bias_n = belds[n] + Nirow[n];
            const float qn     = Qrow[n];
            #pragma unroll
            for (int v = 0; v < 8; ++v) {
                const int   jr  = mb + v;             // row within tile
                const float val = c[nt][v] + bias_n + Njrow0[jr * GDH + n];
                __builtin_nontemporal_store(val, &e_out[(r0 + jr) * GDH + n]);
                pv[v] += val * qn * Krow0[jr * GDH + n];
            }
        }

        // Reduce pv over the 16-lane C group (xor masks < 16 stay in-group).
        #pragma unroll
        for (int s = 1; s < 16; s <<= 1) {
            #pragma unroll
            for (int v = 0; v < 8; ++v) pv[v] += __shfl_xor(pv[v], s, 32);
        }

        // Lanes (m < 8) in each half write one att row: att stored [b][j][i].
        if (m < 8) {
            float val = pv[0];
            #pragma unroll
            for (int q = 1; q < 8; ++q) if (m == q) val = pv[q];
            const int j = j0 + mb + m;
            attT[(long)(b * GN + j) * GN + i] = val * 0.125f;  // 1/sqrt(64)
        }
    }
}

// ---------------------------------------------------------------------------
// Kernel 3: softmax over i (axis 1) for each (b,j) column of attT.
// Writes normalized probs in [b][i][j] layout for kernel 4.
// ---------------------------------------------------------------------------
__global__ __launch_bounds__(256) void softmax_kernel(
    const float* __restrict__ attT, float* __restrict__ attP)
{
    __shared__ float red[256];
    const int bj = blockIdx.x;                 // b*N + j
    const int t  = threadIdx.x;
    const float* col = attT + (long)bj * GN;   // contiguous in i

    const float x0 = col[t];
    const float x1 = col[t + 256];
    red[t] = fmaxf(x0, x1);
    __syncthreads();
    for (int s = 128; s > 0; s >>= 1) {
        if (t < s) red[t] = fmaxf(red[t], red[t + s]);
        __syncthreads();
    }
    const float mx = red[0];
    __syncthreads();

    const float e0 = __expf(x0 - mx);
    const float e1 = __expf(x1 - mx);
    red[t] = e0 + e1;
    __syncthreads();
    for (int s = 128; s > 0; s >>= 1) {
        if (t < s) red[t] += red[t + s];
        __syncthreads();
    }
    const float inv = 1.0f / red[0];

    const int b = bj >> 9, j = bj & (GN - 1);
    attP[((long)(b * GN + t)       * GN) + j] = e0 * inv;  // i = t
    attP[((long)(b * GN + t + 256) * GN) + j] = e1 * inv;  // i = t + 256
}

// ---------------------------------------------------------------------------
// Kernel 4: h_out[b,i,d] = sum_j attP[b,i,j] * V[b,j,d]   (134 MFLOP, tiny)
// ---------------------------------------------------------------------------
__global__ __launch_bounds__(64) void av_kernel(
    const float* __restrict__ attP, const float* __restrict__ V,
    float* __restrict__ h_out)
{
    const int bi = blockIdx.x;                 // b*N + i
    const int d  = threadIdx.x;                // 0..63
    const int b  = bi >> 9;
    const float* arow = attP + (long)bi * GN;
    const float* vb   = V + (long)(b * GN) * GDH;
    float acc = 0.f;
    #pragma unroll 4
    for (int j = 0; j < GN; ++j) acc += arow[j] * vb[j * GDH + d];
    h_out[(long)bi * GDH + d] = acc;
}

// ---------------------------------------------------------------------------
extern "C" void kernel_launch(void* const* d_in, const int* in_sizes, int n_in,
                              void* d_out, int out_size, void* d_ws, size_t ws_size,
                              hipStream_t stream)
{
    const float* h   = (const float*)d_in[0];
    const float* e   = (const float*)d_in[1];
    const float* Wq  = (const float*)d_in[2];  const float* bq  = (const float*)d_in[3];
    const float* Wk  = (const float*)d_in[4];  const float* bk  = (const float*)d_in[5];
    const float* Wv  = (const float*)d_in[6];  const float* bv  = (const float*)d_in[7];
    const float* We  = (const float*)d_in[8];  const float* be  = (const float*)d_in[9];
    const float* Wni = (const float*)d_in[10]; const float* bni = (const float*)d_in[11];
    const float* Wnj = (const float*)d_in[12]; const float* bnj = (const float*)d_in[13];

    float* out   = (float*)d_out;
    float* h_out = out;                               // [B*N*DH]
    float* e_out = out + (long)GB * GN * GDH;         // [B*N*N*DH]

    // Workspace layout (floats): Q,K,V,Ni,Nj (5 x 131072), attT, attP (2 x 1M)
    float* ws   = (float*)d_ws;
    const long SZ = (long)GB * GN * GDH;              // 131072
    float* Qb   = ws;
    float* Kb   = ws + SZ;
    float* Vb   = ws + 2 * SZ;
    float* Nib  = ws + 3 * SZ;
    float* Njb  = ws + 4 * SZ;
    float* attT = ws + 5 * SZ;
    float* attP = attT + (long)GB * GN * GN;

    proj_kernel<<<GB * GN, 320, 0, stream>>>(h, Wq, bq, Wk, bk, Wv, bv,
                                             Wni, bni, Wnj, bnj, Qb);
    egemm_kernel<<<ROWS / 16 / 8 / REPS, 256, 0, stream>>>(e, We, be, Qb, Kb,
                                                           Nib, Njb, e_out, attT);
    softmax_kernel<<<GB * GN, 256, 0, stream>>>(attT, attP);
    av_kernel<<<GB * GN, 64, 0, stream>>>(attP, Vb, h_out);
}
